// SlotAttention_566935683226
// MI455X (gfx1250) — matrix operand order repaired and checked by reading, not compiled, not run on hardware
//
#include <hip/hip_runtime.h>
#include <hip/hip_bf16.h>

// ---------------------------------------------------------------------------
// Slot Attention for MI455X (gfx1250, wave32, WMMA).
// All GEMMs run on v_wmma_f32_16x16x32_f16 (f16 A/B, fp32 accumulate).
// Elementwise math (LN, softmax, GRU gates, residual) stays fp32.
// ---------------------------------------------------------------------------

typedef _Float16 half_t;
typedef __attribute__((ext_vector_type(8)))  _Float16 v8h;
typedef __attribute__((ext_vector_type(16))) _Float16 v16h;
typedef __attribute__((ext_vector_type(8)))  float    v8f;

union F16x16 { v16h v; v8h h[2]; _Float16 e[16]; };
union F32x8  { v8f  v; float e[8]; };

#define BN 64
#define NN 1024
#define DN 768
#define KS 8
#define HN 3072
#define LN_EPS 1e-5f
#define ATT_EPS 1e-8f

// ---------------------------------------------------------------------------
// Generic batched GEMM:  C[m,n] = scale * sum_k A[m,k] * B[n,k]  (+ bias[n])
// A: [M,Kd] f16 (lda), B: [N,Kd] f16 (ldb)  -> C fp32 and/or Ch f16 (ldc).
// One 128-thread block (4 waves) computes a 64x64 tile; each wave owns a
// 16(M) x 64(N) strip as 4 WMMA accumulators. Fragment layouts follow
// CDNA5 ISA 7.12.2 (16-bit A 16x32, B 32x16, 32-bit C/D 16x16).
// All fragment loads of a K-step are issued before the 4 WMMAs so the
// scheduler can overlap WMMA with in-flight loads (partial loadcnt waits).
// MGUARD=false variant (M % 64 == 0) has no EXEC divergence in the hot loop;
// MGUARD=true supports ragged M (the M=8 per-batch slot GEMMs).
// N, Kd must be multiples of 64 / 32 (true for 768/1024/2304/3072).
// ---------------------------------------------------------------------------
template <bool MGUARD>
__global__ void __launch_bounds__(128) gemm_bt_wmma(
    const half_t* __restrict__ A, long long strA,
    const half_t* __restrict__ Bm, long long strB,
    float* __restrict__ C, long long strC,
    half_t* __restrict__ Ch, long long strCh,
    int M, int N, int Kd, int lda, int ldb, int ldc,
    const float* __restrict__ bias, float scale, int relu)
{
    const int bz = blockIdx.z;
    A  += (long long)bz * strA;
    Bm += (long long)bz * strB;

    const int lane = threadIdx.x & 31;
    const int wave = threadIdx.x >> 5;
    const int m0 = blockIdx.y * 64 + wave * 16;
    const int n0 = blockIdx.x * 64;
    if (MGUARD && m0 >= M) return;    // uniform per wave: EXEC stays all-ones

    F32x8 acc[4] = {};

    // A fragment addressing (16x32 f16): lanes 0-15 -> K {0..7,16..23},
    // lanes 16-31 -> K {8..15,24..31}; row = m0 + (lane&15).
    const int arow  = m0 + (lane & 15);
    const bool aval = !MGUARD || (arow < M);
    const int akoff = (lane >> 4) * 8;
    const half_t* arp = A + (long long)arow * lda;

    // B fragment addressing (32x16 f16): col = lane&15,
    // lanes 0-15 -> K {0..7,16..23}, lanes 16-31 -> K {8..15,24..31} (same
    // order as A; B is row-major [N,Kd] and we compute A*B^T).
    const int bcol  = lane & 15;
    const int bkoff = (lane >> 4) * 8;
    const half_t* bp0 = Bm + (long long)(n0 + bcol) * ldb + bkoff;

    for (int k0 = 0; k0 < Kd; k0 += 32) {
        // Prefetch the A stream (HBM-bound operand) two K-steps ahead.
        // Unconditional: worst case lands in the adjacent workspace buffer;
        // prefetches are speculative and side-effect free.
        __builtin_prefetch(arp + k0 + 64, 0, 1);

        // Issue ALL fragment loads for this K-step first...
        F16x16 af;
        if (MGUARD) {
            if (aval) {
                af.h[0] = *(const v8h*)(arp + k0 + akoff);
                af.h[1] = *(const v8h*)(arp + k0 + 16 + akoff);
            } else {
                #pragma unroll
                for (int i = 0; i < 16; ++i) af.e[i] = (_Float16)0.f;
            }
        } else {
            af.h[0] = *(const v8h*)(arp + k0 + akoff);
            af.h[1] = *(const v8h*)(arp + k0 + 16 + akoff);
        }
        F16x16 bf[4];
        #pragma unroll
        for (int f = 0; f < 4; ++f) {
            const half_t* bp = bp0 + (long long)(16 * f) * ldb + k0;
            bf[f].h[0] = *(const v8h*)(bp);
            bf[f].h[1] = *(const v8h*)(bp + 16);
        }
        // ...then the 4 WMMAs, letting later loads stay in flight.
        #pragma unroll
        for (int f = 0; f < 4; ++f)
            acc[f].v = __builtin_amdgcn_wmma_f32_16x16x32_f16(
                false, af.v, false, bf[f].v, (short)0, acc[f].v, false, false);
    }

    if (C)  C  += (long long)bz * strC;
    if (Ch) Ch += (long long)bz * strCh;

    // C layout: lane 0-15 VGPR r -> row m0+r; lane 16-31 VGPR r -> row m0+8+r.
    const int rbase = m0 + ((lane >> 4) << 3);
    #pragma unroll
    for (int f = 0; f < 4; ++f) {
        const int col = n0 + 16 * f + (lane & 15);
        const float bv = bias ? bias[col] : 0.f;
        #pragma unroll
        for (int r = 0; r < 8; ++r) {
            const int row = rbase + r;
            if (!MGUARD || row < M) {
                float v = acc[f].e[r] * scale + bv;
                if (relu) v = fmaxf(v, 0.f);
                if (C)  C [(long long)row * ldc + col] = v;
                if (Ch) Ch[(long long)row * ldc + col] = (half_t)v;
            }
        }
    }
}

// ---------------------------------------------------------------------------
// Row LayerNorm (fp32 in -> f16 out), one 256-thread block per row.
// ---------------------------------------------------------------------------
__global__ void __launch_bounds__(256) ln_rows_f16(
    const float* __restrict__ X, const float* __restrict__ g,
    const float* __restrict__ b, half_t* __restrict__ Y, int D)
{
    const long long row = blockIdx.x;
    const float* x = X + row * D;
    half_t* y = Y + row * D;
    __shared__ float red[256];
    const int tid = threadIdx.x;
    const float invD = 1.0f / (float)D;

    float s = 0.f, ss = 0.f;
    for (int i = tid; i < D; i += 256) { float v = x[i]; s += v; ss += v * v; }

    red[tid] = s; __syncthreads();
    for (int o = 128; o > 0; o >>= 1) { if (tid < o) red[tid] += red[tid + o]; __syncthreads(); }
    const float mean = red[0] * invD;
    __syncthreads();
    red[tid] = ss; __syncthreads();
    for (int o = 128; o > 0; o >>= 1) { if (tid < o) red[tid] += red[tid + o]; __syncthreads(); }
    const float var = red[0] * invD - mean * mean;
    const float inv = rsqrtf(var + LN_EPS);

    for (int i = tid; i < D; i += 256)
        y[i] = (half_t)(((x[i] - mean) * inv) * g[i] + b[i]);
}

// ---------------------------------------------------------------------------
// Slot-axis softmax + row normalization. One block per batch b.
// logits [B,8,N] fp32 -> attn fp32 (softmax over slots, written to d_out)
//                      -> attn_n f16 (attn / (sum_n attn + eps)) for WMMA.
// Whole 8x1024 tile kept in LDS (32KB << 320KB WGP LDS).
// ---------------------------------------------------------------------------
__global__ void __launch_bounds__(256) softmax_slots(
    const float* __restrict__ logits, float* __restrict__ attn_out,
    half_t* __restrict__ attn_h, int N)
{
    const long long b = blockIdx.x;
    const float* L = logits + b * KS * N;
    float*  AO = attn_out + b * KS * N;
    half_t* AH = attn_h   + b * KS * N;

    __shared__ float att[KS * NN];
    __shared__ float rowsum[KS];
    const int tid = threadIdx.x;
    if (tid < KS) rowsum[tid] = 0.f;
    __syncthreads();

    float part[KS];
    #pragma unroll
    for (int k = 0; k < KS; ++k) part[k] = 0.f;

    for (int n = tid; n < N; n += 256) {
        float v[KS];
        float mx = -3.4e38f;
        #pragma unroll
        for (int k = 0; k < KS; ++k) { v[k] = L[k * N + n]; mx = fmaxf(mx, v[k]); }
        float s = 0.f;
        #pragma unroll
        for (int k = 0; k < KS; ++k) { v[k] = __expf(v[k] - mx); s += v[k]; }
        const float inv = 1.0f / s;
        #pragma unroll
        for (int k = 0; k < KS; ++k) {
            const float a = v[k] * inv;
            att[k * NN + n] = a;
            AO[k * N + n] = a;        // softmax (pre-normalization) is returned
            part[k] += a;
        }
    }
    #pragma unroll
    for (int k = 0; k < KS; ++k) atomicAdd(&rowsum[k], part[k]);   // ds_add_f32
    __syncthreads();

    float rinv[KS];
    #pragma unroll
    for (int k = 0; k < KS; ++k) rinv[k] = 1.0f / (rowsum[k] + ATT_EPS);

    for (int n = tid; n < N; n += 256)
        #pragma unroll
        for (int k = 0; k < KS; ++k)
            AH[k * N + n] = (half_t)(att[k * NN + n] * rinv[k]);
}

// ---------------------------------------------------------------------------
// GRU gate fusion: gi/gh [rows, 3D] fp32, h = (1-z)*n + z*h_prev.
// ---------------------------------------------------------------------------
__global__ void __launch_bounds__(256) gru_combine(
    const float* __restrict__ gi, const float* __restrict__ gh,
    const float* __restrict__ sprev, float* __restrict__ h, int rows, int D)
{
    const long long i = (long long)blockIdx.x * blockDim.x + threadIdx.x;
    if (i >= (long long)rows * D) return;
    const int r = (int)(i / D), c = (int)(i % D);
    const float* gir = gi + (long long)r * 3 * D;
    const float* ghr = gh + (long long)r * 3 * D;
    const float ir = gir[c], iz = gir[D + c], in_ = gir[2 * D + c];
    const float hr = ghr[c], hz = ghr[D + c], hn  = ghr[2 * D + c];
    const float rr = 1.f / (1.f + __expf(-(ir + hr)));
    const float zz = 1.f / (1.f + __expf(-(iz + hz)));
    const float nn = tanhf(in_ + rr * hn);
    h[i] = (1.f - zz) * nn + zz * sprev[i];
}

// ---------------------------------------------------------------------------
// Small elementwise helpers.
// ---------------------------------------------------------------------------
__global__ void __launch_bounds__(256) slots_init_k(
    const float* __restrict__ mu, const float* __restrict__ ls,
    const float* __restrict__ noise, float* __restrict__ slots, int D, long long n)
{
    const long long i = (long long)blockIdx.x * blockDim.x + threadIdx.x;
    if (i < n) { const int c = (int)(i % D); slots[i] = mu[c] + __expf(ls[c]) * noise[i]; }
}

__global__ void __launch_bounds__(256) f32_to_f16_k(
    const float* __restrict__ s, half_t* __restrict__ d, long long n)
{
    const long long i = (long long)blockIdx.x * blockDim.x + threadIdx.x;
    if (i < n) d[i] = (half_t)s[i];
}

__global__ void __launch_bounds__(256) add2_k(
    const float* __restrict__ a, const float* __restrict__ b,
    float* __restrict__ o, long long n)
{
    const long long i = (long long)blockIdx.x * blockDim.x + threadIdx.x;
    if (i < n) o[i] = a[i] + b[i];
}

// ---------------------------------------------------------------------------
// f16 transpose per batch: src [R,Cc] -> dst [Cc,R], 32x32 LDS tiles.
// ---------------------------------------------------------------------------
__global__ void __launch_bounds__(256) transpose_f16(
    const half_t* __restrict__ src, half_t* __restrict__ dst, int R, int Cc)
{
    __shared__ half_t tile[32][33];
    const long long bz = blockIdx.z;
    const half_t* S = src + bz * (long long)R * Cc;
    half_t*      Dd = dst + bz * (long long)R * Cc;
    const int c0 = blockIdx.x * 32, r0 = blockIdx.y * 32;
    const int tx = threadIdx.x & 31, ty = threadIdx.x >> 5;
    #pragma unroll
    for (int i = 0; i < 32; i += 8)
        tile[ty + i][tx] = S[(long long)(r0 + ty + i) * Cc + c0 + tx];
    __syncthreads();
    #pragma unroll
    for (int i = 0; i < 32; i += 8)
        Dd[(long long)(c0 + ty + i) * R + r0 + tx] = tile[tx][ty + i];
}

// ---------------------------------------------------------------------------
// Host orchestration.
// ---------------------------------------------------------------------------
static void launch_gemm(hipStream_t s,
    const half_t* A, long long sA, const half_t* Bm, long long sB,
    float* C, long long sC, half_t* Ch, long long sCh,
    int M, int N, int Kd, int lda, int ldb, int ldc,
    const float* bias, float scale, int relu, int batch)
{
    dim3 grid(N / 64, (M + 63) / 64, batch);
    if (M % 64 == 0)
        gemm_bt_wmma<false><<<grid, dim3(128), 0, s>>>(A, sA, Bm, sB, C, sC, Ch, sCh,
                                                       M, N, Kd, lda, ldb, ldc, bias, scale, relu);
    else
        gemm_bt_wmma<true><<<grid, dim3(128), 0, s>>>(A, sA, Bm, sB, C, sC, Ch, sCh,
                                                      M, N, Kd, lda, ldb, ldc, bias, scale, relu);
}

extern "C" void kernel_launch(void* const* d_in, const int* in_sizes, int n_in,
                              void* d_out, int out_size, void* d_ws, size_t ws_size,
                              hipStream_t stream)
{
    (void)in_sizes; (void)n_in; (void)out_size; (void)ws_size;

    const float* inputs = (const float*)d_in[0];
    const float* noise  = (const float*)d_in[1];
    const float* s_mu   = (const float*)d_in[2];
    const float* s_ls   = (const float*)d_in[3];
    const float* lin_g  = (const float*)d_in[4];
    const float* lin_b  = (const float*)d_in[5];
    const float* lsl_g  = (const float*)d_in[6];
    const float* lsl_b  = (const float*)d_in[7];
    const float* q_w    = (const float*)d_in[8];
    const float* k_w    = (const float*)d_in[9];
    const float* v_w    = (const float*)d_in[10];
    const float* w_ih   = (const float*)d_in[11];
    const float* w_hh   = (const float*)d_in[12];
    const float* b_ih   = (const float*)d_in[13];
    const float* b_hh   = (const float*)d_in[14];
    const float* lm_g   = (const float*)d_in[15];
    const float* lm_b   = (const float*)d_in[16];
    const float* w1     = (const float*)d_in[17];
    const float* b1     = (const float*)d_in[18];
    const float* w2     = (const float*)d_in[19];
    const float* b2     = (const float*)d_in[20];

    float* slots    = (float*)d_out;                           // [64,8,768]
    float* attn_out = (float*)d_out + (long long)BN * KS * DN; // [64,8,1024]

    // ---- workspace carve-up (256B aligned) ----
    size_t off = 0;
    char* base = (char*)d_ws;
    auto alloc = [&](size_t bytes) -> char* {
        char* p = base + off; off += (bytes + 255) & ~(size_t)255; return p;
    };
    const long long ND = (long long)BN * NN * DN;   // 50.3M
    const long long SD = (long long)BN * KS * DN;   // 393K  (512 x 768)
    const long long SN = (long long)BN * KS * NN;   // 512K  (512 x 1024)

    half_t* x_h    = (half_t*)alloc(ND * 2);
    half_t* k_h    = (half_t*)alloc(ND * 2);
    half_t* v_h    = (half_t*)alloc(ND * 2);
    half_t* vT_h   = (half_t*)alloc(ND * 2);
    half_t* qw_h   = (half_t*)alloc((size_t)DN * DN * 2);
    half_t* kw_h   = (half_t*)alloc((size_t)DN * DN * 2);
    half_t* vw_h   = (half_t*)alloc((size_t)DN * DN * 2);
    half_t* wih_h  = (half_t*)alloc((size_t)3 * DN * DN * 2);
    half_t* whh_h  = (half_t*)alloc((size_t)3 * DN * DN * 2);
    half_t* w1_h   = (half_t*)alloc((size_t)HN * DN * 2);
    half_t* w2_h   = (half_t*)alloc((size_t)DN * HN * 2);
    half_t* sprev_h= (half_t*)alloc(SD * 2);
    half_t* slln_h = (half_t*)alloc(SD * 2);
    half_t* q_h    = (half_t*)alloc(SD * 2);
    half_t* upd_h  = (half_t*)alloc(SD * 2);
    half_t* y_h    = (half_t*)alloc(SD * 2);
    half_t* hid_h  = (half_t*)alloc((size_t)BN * KS * HN * 2);
    half_t* attn_h = (half_t*)alloc(SN * 2);
    float*  logits = (float*) alloc(SN * 4);
    float*  gi_f   = (float*) alloc((size_t)BN * KS * 3 * DN * 4);
    float*  gh_f   = (float*) alloc((size_t)BN * KS * 3 * DN * 4);
    float*  h_f    = (float*) alloc(SD * 4);
    float*  out2_f = (float*) alloc(SD * 4);

    const float SCALE = 1.0f / sqrtf((float)DN);
    auto cvt = [&](const float* s, half_t* d, long long n) {
        f32_to_f16_k<<<dim3((unsigned)((n + 255) / 256)), dim3(256), 0, stream>>>(s, d, n);
    };

    // ---- one-time (per call) conversions ----
    cvt(q_w,  qw_h,  (long long)DN * DN);
    cvt(k_w,  kw_h,  (long long)DN * DN);
    cvt(v_w,  vw_h,  (long long)DN * DN);
    cvt(w_ih, wih_h, (long long)3 * DN * DN);
    cvt(w_hh, whh_h, (long long)3 * DN * DN);
    cvt(w1,   w1_h,  (long long)HN * DN);
    cvt(w2,   w2_h,  (long long)DN * HN);

    // x = LN(inputs) -> f16
    ln_rows_f16<<<dim3(BN * NN), dim3(256), 0, stream>>>(inputs, lin_g, lin_b, x_h, DN);

    // k = x @ k_w^T ; v = x @ v_w^T   (f16 out; memory-bound — x_h fits in L2)
    launch_gemm(stream, x_h, 0, kw_h, 0, nullptr, 0, k_h, 0,
                BN * NN, DN, DN, DN, DN, DN, nullptr, 1.f, 0, 1);
    launch_gemm(stream, x_h, 0, vw_h, 0, nullptr, 0, v_h, 0,
                BN * NN, DN, DN, DN, DN, DN, nullptr, 1.f, 0, 1);

    // vT[b] = v[b]^T  ([1024,768] -> [768,1024]) for the updates GEMM
    transpose_f16<<<dim3(DN / 32, NN / 32, BN), dim3(256), 0, stream>>>(v_h, vT_h, NN, DN);

    // slots = mu + exp(log_sigma) * noise   (lives in d_out)
    slots_init_k<<<dim3((unsigned)((SD + 255) / 256)), dim3(256), 0, stream>>>(
        s_mu, s_ls, noise, slots, DN, SD);

    for (int it = 0; it < 3; ++it) {
        // slots_prev (f16 copy for GRU hidden GEMM)
        cvt(slots, sprev_h, SD);

        // q = LN(slots) @ q_w^T
        ln_rows_f16<<<dim3(BN * KS), dim3(256), 0, stream>>>(slots, lsl_g, lsl_b, slln_h, DN);
        launch_gemm(stream, slln_h, 0, qw_h, 0, nullptr, 0, q_h, 0,
                    BN * KS, DN, DN, DN, DN, DN, nullptr, 1.f, 0, 1);

        // logits[b] = SCALE * q[b] @ k[b]^T   (batched, M=8 guarded)
        launch_gemm(stream, q_h, (long long)KS * DN, k_h, (long long)NN * DN,
                    logits, (long long)KS * NN, nullptr, 0,
                    KS, NN, DN, DN, DN, NN, nullptr, SCALE, 0, BN);

        // attn = softmax over slots; attn_n = attn / (row-sum + eps)
        softmax_slots<<<dim3(BN), dim3(256), 0, stream>>>(logits, attn_out, attn_h, NN);

        // updates[b] = attn_n[b] @ v[b]  ==  attn_n[b] @ vT[b]^T  (batched)
        launch_gemm(stream, attn_h, (long long)KS * NN, vT_h, (long long)DN * NN,
                    nullptr, 0, upd_h, (long long)KS * DN,
                    KS, DN, NN, NN, NN, DN, nullptr, 1.f, 0, BN);

        // GRU: gi = updates @ w_ih^T + b_ih ; gh = slots_prev @ w_hh^T + b_hh
        launch_gemm(stream, upd_h, 0, wih_h, 0, gi_f, 0, nullptr, 0,
                    BN * KS, 3 * DN, DN, DN, DN, 3 * DN, b_ih, 1.f, 0, 1);
        launch_gemm(stream, sprev_h, 0, whh_h, 0, gh_f, 0, nullptr, 0,
                    BN * KS, 3 * DN, DN, DN, DN, 3 * DN, b_hh, 1.f, 0, 1);
        gru_combine<<<dim3((unsigned)((SD + 255) / 256)), dim3(256), 0, stream>>>(
            gi_f, gh_f, slots, h_f, BN * KS, DN);

        // MLP: slots = h + relu(LN(h) @ w1^T + b1) @ w2^T + b2
        ln_rows_f16<<<dim3(BN * KS), dim3(256), 0, stream>>>(h_f, lm_g, lm_b, y_h, DN);
        launch_gemm(stream, y_h, 0, w1_h, 0, nullptr, 0, hid_h, 0,
                    BN * KS, HN, DN, DN, DN, HN, b1, 1.f, 1, 1);
        launch_gemm(stream, hid_h, 0, w2_h, 0, out2_f, 0, nullptr, 0,
                    BN * KS, DN, HN, HN, HN, DN, b2, 1.f, 0, 1);
        add2_k<<<dim3((unsigned)((SD + 255) / 256)), dim3(256), 0, stream>>>(
            h_f, out2_f, slots, SD);
    }
}
